// MoEBlock_45019847197309
// MI455X (gfx1250) — matrix operand order repaired and checked by reading, not compiled, not run on hardware
//
#include <hip/hip_runtime.h>

// ---------------------------------------------------------------------------
// MoE block (top-2 of 8 experts, SwiGLU) for gfx1250 using bf16 WMMA.
// T=4096 tokens, H=1024, I=2048, E=8.
// One-shot fp32->bf16 transpose-convert of weights, then grouped GEMMs.
// Tiles are staged with CDNA5 GLOBAL_LOAD_ASYNC_TO_LDS_B128 (ASYNCcnt):
// no VGPR staging, no spills; DMA of tile k+1 overlaps WMMA of tile k.
// ---------------------------------------------------------------------------

#define T_TOK 4096
#define HDIM  1024
#define IDIM  2048
#define NEXP  8

// LDS row stride (in ushorts) for 64-wide K tiles: 72*2 = 144 B = 9*16 B,
// keeps every 16B access aligned and staggers banks.
#define XST 72

#define AS3 __attribute__((address_space(3)))

typedef __attribute__((ext_vector_type(16))) __bf16          v16bf;
typedef __attribute__((ext_vector_type(8)))  float           v8f;
typedef __attribute__((ext_vector_type(8)))  unsigned short  us8;

union Frag {
  v16bf bf;
  us8   u[2];
};

__device__ __forceinline__ unsigned short f2bf(float f) {
  unsigned int u = __float_as_uint(f);
  unsigned int r = u + 0x7FFFu + ((u >> 16) & 1u);   // round-to-nearest-even
  return (unsigned short)(r >> 16);
}

__device__ __forceinline__ unsigned int pk2bf(float lo, float hi) {
  return (unsigned int)f2bf(lo) | ((unsigned int)f2bf(hi) << 16);
}

// CDNA5 async global->LDS copy, 16 bytes per lane (ISA §15.18.3 op 98).
// dst: 32-bit LDS address (addrspace(3)); src: 64-bit global address.
__device__ __forceinline__ void async_ld_b128(AS3 unsigned short* dst,
                                              const unsigned short* src) {
  asm volatile("global_load_async_to_lds_b128 %0, %1, off"
               :: "v"(dst), "v"(src) : "memory");
}
__device__ __forceinline__ void wait_async0() {
  asm volatile("s_wait_asynccnt 0x0" ::: "memory");
}

// ---------------------------------------------------------------------------
__global__ void zero_counts_kernel(int* cnt) {
  if (threadIdx.x < NEXP) cnt[threadIdx.x] = 0;
}

// float32 x -> bf16 xb, vectorized by 4
__global__ __launch_bounds__(256) void cvtx_kernel(const float* __restrict__ x,
                                                   unsigned short* __restrict__ xb,
                                                   int n4) {
  int i = blockIdx.x * blockDim.x + threadIdx.x;
  if (i >= n4) return;
  float4 v = ((const float4*)x)[i];
  unsigned int lo = pk2bf(v.x, v.y);
  unsigned int hi = pk2bf(v.z, v.w);
  unsigned long long packed = (unsigned long long)lo | ((unsigned long long)hi << 32);
  ((unsigned long long*)xb)[i] = packed;
}

// ---------------------------------------------------------------------------
// Transpose + convert: src [E][R][C] f32 -> dst [E][C][R] bf16.
// 64x64 tiles through LDS; coalesced f32 reads, coalesced 16B bf16 writes.
__global__ __launch_bounds__(256) void tconv_kernel(const float* __restrict__ src,
                                                    unsigned short* __restrict__ dst,
                                                    int R, int C) {
  const size_t sbase = (size_t)blockIdx.z * R * C;
  const int r0 = blockIdx.y * 64;
  const int c0 = blockIdx.x * 64;
  __shared__ __align__(16) unsigned short tile[64][XST];
  const int tid = threadIdx.x;
#pragma unroll
  for (int i = 0; i < 4; ++i) {
    int idx = tid + i * 256;                  // 1024 float4 jobs
    int rr = idx >> 4;                        // 0..63
    int cc = (idx & 15) * 4;                  // 0..60
    float4 v = *(const float4*)(src + sbase + (size_t)(r0 + rr) * C + c0 + cc);
    tile[cc + 0][rr] = f2bf(v.x);
    tile[cc + 1][rr] = f2bf(v.y);
    tile[cc + 2][rr] = f2bf(v.z);
    tile[cc + 3][rr] = f2bf(v.w);
  }
  __syncthreads();
#pragma unroll
  for (int i = 0; i < 2; ++i) {
    int idx = tid + i * 256;                  // 512 uint4 jobs
    int cc  = idx >> 3;                       // 0..63 (dst row = src col)
    int seg = idx & 7;
    uint4 v = *(const uint4*)&tile[cc][seg * 8];
    *(uint4*)(dst + sbase + (size_t)(c0 + cc) * R + r0 + seg * 8) = v;
  }
}

// ---------------------------------------------------------------------------
// Router: one wave (32 lanes) per token. Computes x[t] @ Wr (8 logits),
// top-2 + softmax, and builds per-expert token-slot lists with atomics.
__global__ __launch_bounds__(256) void router_kernel(const float* __restrict__ x,
                                                     const float* __restrict__ Wr,
                                                     float* __restrict__ wbuf,
                                                     int*   __restrict__ list,
                                                     int*   __restrict__ cnt) {
  int wave = blockIdx.x * (blockDim.x >> 5) + (threadIdx.x >> 5);
  int lane = threadIdx.x & 31;
  if (wave >= T_TOK) return;
  const float* xr = x + (size_t)wave * HDIM;
  float acc[NEXP];
#pragma unroll
  for (int e = 0; e < NEXP; ++e) acc[e] = 0.f;

  for (int h0 = lane * 4; h0 < HDIM; h0 += 128) {
    float4 xv = *(const float4*)(xr + h0);
#pragma unroll
    for (int j = 0; j < 4; ++j) {
      float xs = (&xv.x)[j];
      const float* wr = Wr + (size_t)(h0 + j) * NEXP;
      float4 w0 = *(const float4*)(wr);
      float4 w1 = *(const float4*)(wr + 4);
      acc[0] += xs * w0.x; acc[1] += xs * w0.y; acc[2] += xs * w0.z; acc[3] += xs * w0.w;
      acc[4] += xs * w1.x; acc[5] += xs * w1.y; acc[6] += xs * w1.z; acc[7] += xs * w1.w;
    }
  }
#pragma unroll
  for (int e = 0; e < NEXP; ++e) {
#pragma unroll
    for (int off = 16; off >= 1; off >>= 1)
      acc[e] += __shfl_xor(acc[e], off, 32);
  }
  if (lane == 0) {
    int e0 = 0;
#pragma unroll
    for (int e = 1; e < NEXP; ++e) if (acc[e] > acc[e0]) e0 = e;
    int e1 = -1;
#pragma unroll
    for (int e = 0; e < NEXP; ++e) {
      if (e == e0) continue;
      if (e1 < 0 || acc[e] > acc[e1]) e1 = e;
    }
    float v1 = acc[e1] - acc[e0];
    float ex1 = __expf(v1);                     // v0 is the max; ex0 = 1
    float inv = 1.f / (1.f + ex1);
    int t = wave;
    int s0 = atomicAdd(&cnt[e0], 1);
    int s1 = atomicAdd(&cnt[e1], 1);
    list[e0 * T_TOK + s0] = t * 2;
    list[e1 * T_TOK + s1] = t * 2 + 1;
    wbuf[t * 2]     = inv;
    wbuf[t * 2 + 1] = ex1 * inv;
  }
}

// ---------------------------------------------------------------------------
// GEMM1: per expert, gather token rows; A = silu(X*Wg) * (X*Wu), bf16 out.
// Weights pre-transposed bf16: WgT/WuT [E][I][H] (n-major, k contiguous).
// Block: 256 thr = 8 waves = 4 row-groups x {G-wave, U-wave}.
// Each wave: 32M x 64N for ONE matrix -> 64 accumulator VGPRs.
// SwiGLU fused via a one-time LDS exchange (aliased over dead X tiles).
__global__ __launch_bounds__(256) void gemm1_kernel(
    const unsigned short* __restrict__ xb,    // [T, H] bf16
    const unsigned short* __restrict__ WgT,   // [E, I, H] bf16
    const unsigned short* __restrict__ WuT,   // [E, I, H] bf16
    const int*   __restrict__ list,
    const int*   __restrict__ cnt,
    unsigned short* __restrict__ Abuf)        // [2T, I] bf16
{
  const int e     = blockIdx.z;
  const int mtile = blockIdx.y;
  const int ntile = blockIdx.x;
  const int cntE  = cnt[e];
  if (mtile * 128 >= cntE) return;

  // Manually partitioned LDS pool (xchg aliases the dead X region at the end)
  __shared__ __align__(16) unsigned char smem[2 * 128 * XST * 2 +  // Xs[2]
                                              2 * 64 * XST * 2 +   // Gs[2]
                                              2 * 64 * XST * 2];   // Us[2]
  unsigned short* XsP = (unsigned short*)smem;                     // [2][128*XST]
  unsigned short* GsP = (unsigned short*)(smem + 2 * 128 * XST * 2);
  unsigned short* UsP = (unsigned short*)(smem + 2 * 128 * XST * 2 + 2 * 64 * XST * 2);
  float*          xchg = (float*)smem;                             // 128*64 f32 = 32 KB
  __shared__ int pairLds[128];

  const int tid = threadIdx.x;
  if (tid < 128) {
    int rm = mtile * 128 + tid;
    pairLds[tid] = (rm < cntE) ? list[e * T_TOK + rm] : -1;
  }
  __syncthreads();

  const int lane   = tid & 31;
  const int wave   = tid >> 5;
  const int wm     = wave >> 1;   // 0..3  (row group: 32 rows each)
  const int matsel = wave & 1;    // 0 = gate, 1 = up

  v8f acc[2][4];
#pragma unroll
  for (int a = 0; a < 2; ++a)
#pragma unroll
    for (int b = 0; b < 4; ++b) acc[a][b] = (v8f)0.f;

  // bf16 transposed weight base: rows are N (I-dim), k contiguous (H-dim)
  const size_t wbase = (size_t)e * IDIM * HDIM + (size_t)(ntile * 64) * HDIM;

  // --- async DMA staging (branchless gather: invalid rows clamp to row 0;
  //     their D rows are discarded in the epilogue, so no zero-fill needed) ---
  auto stageTile = [&](int k0, int pb) {
#pragma unroll
    for (int i = 0; i < 4; ++i) {
      int idx = tid + i * 256;                // 1024 b128 X jobs
      int row = idx >> 3, seg = idx & 7;
      int pr = pairLds[row];
      int pc = pr < 0 ? 0 : pr;
      async_ld_b128((AS3 unsigned short*)&XsP[pb * (128 * XST) + row * XST + seg * 8],
                    xb + (size_t)(pc >> 1) * HDIM + k0 + seg * 8);
    }
#pragma unroll
    for (int r = 0; r < 2; ++r) {
      int idx = tid + r * 256;                // 512 b128 jobs per matrix
      int n = idx >> 3, seg = idx & 7;
      size_t ga = wbase + (size_t)n * HDIM + k0 + seg * 8;
      async_ld_b128((AS3 unsigned short*)&GsP[pb * (64 * XST) + n * XST + seg * 8], WgT + ga);
      async_ld_b128((AS3 unsigned short*)&UsP[pb * (64 * XST) + n * XST + seg * 8], WuT + ga);
    }
  };

  stageTile(0, 0);
  wait_async0();
  __syncthreads();

  for (int k0 = 0; k0 < HDIM; k0 += 64) {
    const int pb = (k0 >> 6) & 1;
    if (k0 + 64 < HDIM) stageTile(k0 + 64, pb ^ 1);   // DMA overlaps WMMA below
    const unsigned short* Xb = XsP + pb * (128 * XST);
    const unsigned short* Bb = (matsel ? UsP : GsP) + pb * (64 * XST);
    // ---- two 32-K WMMA sub-steps (ISA 7.12.2 bf16 layouts) ----
#pragma unroll
    for (int ks = 0; ks < 64; ks += 32) {
      Frag a[2], b[4];
      const int rb = wm * 32 + (lane & 15);
      const int kb = ks + (lane >> 4) * 8;
#pragma unroll
      for (int mt = 0; mt < 2; ++mt) {
        int r = rb + mt * 16;
        a[mt].u[0] = *(const us8*)&Xb[r * XST + kb];
        a[mt].u[1] = *(const us8*)&Xb[r * XST + kb + 16];
      }
      const int kh = ks + (lane >> 4) * 8;
#pragma unroll
      for (int nt = 0; nt < 4; ++nt) {
        int n = nt * 16 + (lane & 15);
        b[nt].u[0] = *(const us8*)&Bb[n * XST + kh];
        b[nt].u[1] = *(const us8*)&Bb[n * XST + kh + 16];
      }
#pragma unroll
      for (int mt = 0; mt < 2; ++mt)
#pragma unroll
        for (int nt = 0; nt < 4; ++nt)
          acc[mt][nt] = __builtin_amdgcn_wmma_f32_16x16x32_bf16(
              false, a[mt].bf, false, b[nt].bf, (short)0, acc[mt][nt], false, false);
    }
    wait_async0();                             // own DMAs done before barrier
    __syncthreads();
  }

  // ---- epilogue: U-waves publish to LDS (aliased), G-waves fuse SwiGLU ----
  if (matsel) {
#pragma unroll
    for (int mt = 0; mt < 2; ++mt)
#pragma unroll
      for (int nt = 0; nt < 4; ++nt)
#pragma unroll
        for (int j = 0; j < 8; ++j) {
          int rowL = wm * 32 + mt * 16 + (lane >> 4) * 8 + j;   // 0..127
          int col  = nt * 16 + (lane & 15);                     // 0..63
          xchg[rowL * 64 + col] = acc[mt][nt][j];
        }
  }
  __syncthreads();
  if (!matsel) {
#pragma unroll
    for (int mt = 0; mt < 2; ++mt)
#pragma unroll
      for (int nt = 0; nt < 4; ++nt)
#pragma unroll
        for (int j = 0; j < 8; ++j) {
          int rowL = wm * 32 + mt * 16 + (lane >> 4) * 8 + j;
          int col  = nt * 16 + (lane & 15);
          int pair = pairLds[rowL];
          if (pair >= 0) {
            float g = acc[mt][nt][j];
            float u = xchg[rowL * 64 + col];
            float aval = (g / (1.f + __expf(-g))) * u;
            Abuf[(size_t)pair * IDIM + ntile * 64 + col] = f2bf(aval);
          }
        }
  }
}

// ---------------------------------------------------------------------------
// GEMM2: Y[pair] = w[pair] * (A[pair] @ Wd[e]);  WdT [E][H][I] bf16.
__global__ __launch_bounds__(256) void gemm2_kernel(
    const unsigned short* __restrict__ Abuf,  // [2T, I] bf16
    const unsigned short* __restrict__ WdT,   // [E, H, I] bf16
    const int*   __restrict__ list,
    const int*   __restrict__ cnt,
    const float* __restrict__ wbuf,           // [2T]
    float*       __restrict__ Ybuf)           // [2T, H] f32
{
  const int e     = blockIdx.z;
  const int mtile = blockIdx.y;
  const int ntile = blockIdx.x;
  const int cntE  = cnt[e];
  if (mtile * 128 >= cntE) return;

  __shared__ __align__(16) unsigned short Xs[2][128 * XST];
  __shared__ __align__(16) unsigned short Ds[2][64 * XST];
  __shared__ int pairLds[128];

  const int tid = threadIdx.x;
  if (tid < 128) {
    int rm = mtile * 128 + tid;
    pairLds[tid] = (rm < cntE) ? list[e * T_TOK + rm] : -1;
  }
  __syncthreads();

  const int lane = tid & 31;
  const int wave = tid >> 5;
  const int wm = wave >> 1;
  const int wn = wave & 1;

  v8f acc[2][2];
#pragma unroll
  for (int a = 0; a < 2; ++a)
#pragma unroll
    for (int b = 0; b < 2; ++b) acc[a][b] = (v8f)0.f;

  const size_t wbase = (size_t)e * HDIM * IDIM + (size_t)(ntile * 64) * IDIM;

  auto stageTile = [&](int k0, int pb) {
#pragma unroll
    for (int i = 0; i < 4; ++i) {
      int idx = tid + i * 256;
      int row = idx >> 3, seg = idx & 7;
      int pr = pairLds[row];
      int pc = pr < 0 ? 0 : pr;
      async_ld_b128((AS3 unsigned short*)&Xs[pb][row * XST + seg * 8],
                    Abuf + (size_t)pc * IDIM + k0 + seg * 8);
    }
#pragma unroll
    for (int r = 0; r < 2; ++r) {
      int idx = tid + r * 256;
      int n = idx >> 3, seg = idx & 7;
      async_ld_b128((AS3 unsigned short*)&Ds[pb][n * XST + seg * 8],
                    WdT + wbase + (size_t)n * IDIM + k0 + seg * 8);
    }
  };

  stageTile(0, 0);
  wait_async0();
  __syncthreads();

  for (int k0 = 0; k0 < IDIM; k0 += 64) {
    const int pb = (k0 >> 6) & 1;
    if (k0 + 64 < IDIM) stageTile(k0 + 64, pb ^ 1);
#pragma unroll
    for (int ks = 0; ks < 64; ks += 32) {
      Frag a[2], bd[2];
      const int rb = wm * 32 + (lane & 15);
      const int kb = ks + (lane >> 4) * 8;
#pragma unroll
      for (int mt = 0; mt < 2; ++mt) {
        int r = rb + mt * 16;
        a[mt].u[0] = *(const us8*)&Xs[pb][r * XST + kb];
        a[mt].u[1] = *(const us8*)&Xs[pb][r * XST + kb + 16];
      }
      const int nb = wn * 32 + (lane & 15);
      const int kh = ks + (lane >> 4) * 8;
#pragma unroll
      for (int nt = 0; nt < 2; ++nt) {
        int n = nb + nt * 16;
        bd[nt].u[0] = *(const us8*)&Ds[pb][n * XST + kh];
        bd[nt].u[1] = *(const us8*)&Ds[pb][n * XST + kh + 16];
      }
#pragma unroll
      for (int mt = 0; mt < 2; ++mt)
#pragma unroll
        for (int nt = 0; nt < 2; ++nt)
          acc[mt][nt] = __builtin_amdgcn_wmma_f32_16x16x32_bf16(
              false, a[mt].bf, false, bd[nt].bf, (short)0, acc[mt][nt], false, false);
    }
    wait_async0();
    __syncthreads();
  }

  const int colBase = ntile * 64 + wn * 32 + (lane & 15);
#pragma unroll
  for (int mt = 0; mt < 2; ++mt) {
    int rowBase = wm * 32 + mt * 16 + (lane >> 4) * 8;
#pragma unroll
    for (int nt = 0; nt < 2; ++nt) {
      int col = colBase + nt * 16;
#pragma unroll
      for (int j = 0; j < 8; ++j) {
        int pair = pairLds[rowBase + j];
        if (pair >= 0)
          Ybuf[(size_t)pair * HDIM + col] = wbuf[pair] * acc[mt][nt][j];
      }
    }
  }
}

// ---------------------------------------------------------------------------
// Combine: out[t] = Y[2t] + Y[2t+1]  (both expert contributions, weighted)
__global__ __launch_bounds__(256) void combine_kernel(const float* __restrict__ Ybuf,
                                                      float* __restrict__ out, int n4) {
  int i = blockIdx.x * blockDim.x + threadIdx.x;
  if (i >= n4) return;
  int t  = i >> 8;           // HDIM/4 = 256 float4 per row
  int h4 = i & 255;
  const float4* y = (const float4*)Ybuf;
  float4 a = y[(size_t)(2 * t) * 256 + h4];
  float4 b = y[(size_t)(2 * t + 1) * 256 + h4];
  float4 o; o.x = a.x + b.x; o.y = a.y + b.y; o.z = a.z + b.z; o.w = a.w + b.w;
  ((float4*)out)[i] = o;
}

// ---------------------------------------------------------------------------
extern "C" void kernel_launch(void* const* d_in, const int* in_sizes, int n_in,
                              void* d_out, int out_size, void* d_ws, size_t ws_size,
                              hipStream_t stream) {
  const float* x  = (const float*)d_in[0];
  const float* Wr = (const float*)d_in[1];
  const float* Wg = (const float*)d_in[2];
  const float* Wu = (const float*)d_in[3];
  const float* Wd = (const float*)d_in[4];
  float* out = (float*)d_out;

  char* ws = (char*)d_ws;
  size_t off = 0;
  auto alloc = [&](size_t bytes) {
    char* p = ws + off;
    off = (off + bytes + 255) & ~(size_t)255;
    return p;
  };
  unsigned short* xb   = (unsigned short*)alloc((size_t)T_TOK * HDIM * 2);       // 8.4 MB
  unsigned short* Abuf = (unsigned short*)alloc((size_t)2 * T_TOK * IDIM * 2);   // 33.5 MB
  float*          Ybuf = (float*)alloc((size_t)2 * T_TOK * HDIM * 4);            // 33.5 MB
  unsigned short* WgT  = (unsigned short*)alloc((size_t)NEXP * HDIM * IDIM * 2); // 33.5 MB
  unsigned short* WuT  = (unsigned short*)alloc((size_t)NEXP * HDIM * IDIM * 2); // 33.5 MB
  unsigned short* WdT  = (unsigned short*)alloc((size_t)NEXP * HDIM * IDIM * 2); // 33.5 MB
  float*          wbuf = (float*)alloc((size_t)2 * T_TOK * 4);
  int*            list = (int*)alloc((size_t)NEXP * T_TOK * 4);
  int*            cnt  = (int*)alloc(NEXP * 4);
  (void)ws_size; (void)in_sizes; (void)n_in; (void)out_size;

  zero_counts_kernel<<<1, 32, 0, stream>>>(cnt);
  cvtx_kernel<<<(T_TOK * HDIM / 4 + 255) / 256, 256, 0, stream>>>(x, xb, T_TOK * HDIM / 4);
  router_kernel<<<T_TOK / 8, 256, 0, stream>>>(x, Wr, wbuf, list, cnt);
  // Wg/Wu: [E][H][I] f32 -> [E][I][H] bf16 ; Wd: [E][I][H] f32 -> [E][H][I] bf16
  tconv_kernel<<<dim3(IDIM / 64, HDIM / 64, NEXP), 256, 0, stream>>>(Wg, WgT, HDIM, IDIM);
  tconv_kernel<<<dim3(IDIM / 64, HDIM / 64, NEXP), 256, 0, stream>>>(Wu, WuT, HDIM, IDIM);
  tconv_kernel<<<dim3(HDIM / 64, IDIM / 64, NEXP), 256, 0, stream>>>(Wd, WdT, IDIM, HDIM);
  gemm1_kernel<<<dim3(IDIM / 64, T_TOK / 128, NEXP), 256, 0, stream>>>(xb, WgT, WuT, list, cnt, Abuf);
  gemm2_kernel<<<dim3(HDIM / 64, T_TOK / 128, NEXP), 256, 0, stream>>>(Abuf, WdT, list, cnt, wbuf, Ybuf);
  combine_kernel<<<(T_TOK * HDIM / 4 + 255) / 256, 256, 0, stream>>>(Ybuf, out, T_TOK * HDIM / 4);
}
